// MLPBigramLanguageModel_15204184227955
// MI455X (gfx1250) — matrix operand order repaired and checked by reading, not compiled, not run on hardware
//
#include <hip/hip_runtime.h>
#include <hip/hip_bf16.h>

#define VOCAB  50257
#define HIDDEN 256
#define BT     4096   // B*T = 4*1024

typedef __bf16 v16bf __attribute__((ext_vector_type(16)));
typedef float  v8f   __attribute__((ext_vector_type(8)));

union Frag16 { uint4 u[2]; v16bf v; };

__device__ __forceinline__ unsigned short f32_to_bf16_rne(float f) {
    unsigned int u = __builtin_bit_cast(unsigned int, f);
    u += 0x7FFFu + ((u >> 16) & 1u);   // round-to-nearest-even
    return (unsigned short)(u >> 16);
}

// ---------------- Kernel 1: W2 fp32 -> bf16 (one row per block) ----------------
__global__ void __launch_bounds__(256)
cvt_w2_kernel(const float* __restrict__ W2, unsigned short* __restrict__ W2b) {
    size_t i = (size_t)blockIdx.x * HIDDEN + threadIdx.x;   // 50257 blocks x 256
    W2b[i] = f32_to_bf16_rne(W2[i]);
}

// ------------- Kernel 2: X = relu(W1.T[idx] + b1) in bf16 [BT, HIDDEN] ---------
__global__ void __launch_bounds__(256)
embed_kernel(const int* __restrict__ idx, const float* __restrict__ W1,
             const float* __restrict__ b1, unsigned short* __restrict__ Xb) {
    int t = blockIdx.x;          // token
    int h = threadIdx.x;         // hidden unit
    int tok = idx[t];
    float v = W1[(size_t)h * VOCAB + tok] + b1[h];
    v = fmaxf(v, 0.0f);
    Xb[(size_t)t * HIDDEN + h] = f32_to_bf16_rne(v);
}

// ------------- Kernel 3: logits = X @ W2^T + b2 via WMMA bf16 ------------------
// Block = 8 waves; wave tile 16(M) x 64(N); block tile 32(M) x 256(N).
// Software-pipelined: next K-step's fragments are in flight while current WMMAs issue.
__global__ void __launch_bounds__(256)
gemm_logits_kernel(const unsigned short* __restrict__ Xb,
                   const unsigned short* __restrict__ W2b,
                   const float* __restrict__ b2,
                   float* __restrict__ out) {
    const int lane = threadIdx.x & 31;
    const int wave = threadIdx.x >> 5;
    const int wm   = wave >> 2;               // 0..1
    const int wn   = wave & 3;                // 0..3
    const int m0   = blockIdx.y * 32 + wm * 16;
    const int n0   = blockIdx.x * 256 + wn * 64;

    const int  lrow = lane & 15;
    const bool hi   = lane >= 16;

    // A fragment (16x32 bf16): lanes0-15 hold K 0-15,
    // lanes16-31 hold K 16-31 of row m0+lrow.
    const uint4* aptr = (const uint4*)Xb + (size_t)(m0 + lrow) * (HIDDEN / 8) + (hi ? 2 : 0);
    // B fragment (32x16 bf16): lane holds one column, contiguous-K half.
    const uint4* bptr[4];
#pragma unroll
    for (int j = 0; j < 4; ++j) {
        int n = n0 + j * 16 + lrow;
        if (n >= VOCAB) n = VOCAB - 1;        // clamp: stores are guarded below
        bptr[j] = (const uint4*)W2b + (size_t)n * (HIDDEN / 8) + (hi ? 2 : 0);
    }

    v8f acc[4] = {};

    // Prologue: load K-step 0 fragments.
    Frag16 a, b[4];
    a.u[0] = aptr[0];
    a.u[1] = aptr[1];
#pragma unroll
    for (int j = 0; j < 4; ++j) { b[j].u[0] = bptr[j][0]; b[j].u[1] = bptr[j][1]; }

#pragma unroll
    for (int kk = 0; kk < HIDDEN; kk += 32) {
        // Issue next K-step loads first (last iter reloads current chunk; harmless).
        const int knext = (kk + 32 < HIDDEN) ? (kk + 32) : kk;
        const int o     = knext >> 3;
        Frag16 an, bn[4];
        an.u[0] = aptr[o];
        an.u[1] = aptr[o + 1];
#pragma unroll
        for (int j = 0; j < 4; ++j) { bn[j].u[0] = bptr[j][o]; bn[j].u[1] = bptr[j][o + 1]; }

        // Compute on current fragments while next loads are in flight.
#pragma unroll
        for (int j = 0; j < 4; ++j) {
            acc[j] = __builtin_amdgcn_wmma_f32_16x16x32_bf16(
                false, a.v, false, b[j].v, (short)0, acc[j], false, false);
        }

        a = an;
#pragma unroll
        for (int j = 0; j < 4; ++j) b[j] = bn[j];
    }

    // D layout (ISA 7.12.2): VGPR e, lanes0-15 -> row m0+e, lanes16-31 -> row m0+8+e.
    // Logits are streaming-once (823MB >> 192MB L2): store non-temporal so the
    // L2-resident bf16 W2 (25.7MB) is not evicted.
#pragma unroll
    for (int j = 0; j < 4; ++j) {
        int n = n0 + j * 16 + lrow;
        if (n < VOCAB) {
            float bias  = b2[n];
            int   rbase = m0 + (hi ? 8 : 0);
#pragma unroll
            for (int e = 0; e < 8; ++e) {
                __builtin_nontemporal_store(acc[j][e] + bias,
                                            &out[(size_t)(rbase + e) * VOCAB + n]);
            }
        }
    }
}

// ------------- Kernel 4: per-token online-softmax LSE + NLL --------------------
__global__ void __launch_bounds__(256)
lse_kernel(const float* __restrict__ logits, const int* __restrict__ targets,
           float* __restrict__ nll) {
    int t = blockIdx.x;
    const float* row = logits + (size_t)t * VOCAB;

    float m = -INFINITY, s = 0.0f;
    for (int v = threadIdx.x; v < VOCAB; v += 256) {
        float x = __builtin_nontemporal_load(&row[v]);   // streaming read, keep L2 for W2
        if (x > m) { s *= expf(m - x); m = x; }
        s += expf(x - m);
    }

    __shared__ float sm[256], ss[256];
    sm[threadIdx.x] = m;
    ss[threadIdx.x] = s;
    __syncthreads();
    for (int off = 128; off > 0; off >>= 1) {
        if ((int)threadIdx.x < off) {
            float m2 = sm[threadIdx.x + off], s2 = ss[threadIdx.x + off];
            float M  = fmaxf(sm[threadIdx.x], m2);
            ss[threadIdx.x] = ss[threadIdx.x] * expf(sm[threadIdx.x] - M) + s2 * expf(m2 - M);
            sm[threadIdx.x] = M;
        }
        __syncthreads();
    }
    if (threadIdx.x == 0) {
        float lse = sm[0] + logf(ss[0]);
        nll[t] = lse - row[targets[t]];
    }
}

// ------------- Kernel 5: loss = mean(nll) --------------------------------------
__global__ void __launch_bounds__(256)
loss_kernel(const float* __restrict__ nll, float* __restrict__ loss_out) {
    __shared__ float sdata[256];
    float s = 0.0f;
    for (int i = threadIdx.x; i < BT; i += 256) s += nll[i];
    sdata[threadIdx.x] = s;
    __syncthreads();
    for (int off = 128; off > 0; off >>= 1) {
        if ((int)threadIdx.x < off) sdata[threadIdx.x] += sdata[threadIdx.x + off];
        __syncthreads();
    }
    if (threadIdx.x == 0) loss_out[0] = sdata[0] / (float)BT;
}

extern "C" void kernel_launch(void* const* d_in, const int* in_sizes, int n_in,
                              void* d_out, int out_size, void* d_ws, size_t ws_size,
                              hipStream_t stream) {
    const int*   idx     = (const int*)d_in[0];
    const int*   targets = (const int*)d_in[1];
    const float* W1      = (const float*)d_in[2];
    const float* b1      = (const float*)d_in[3];
    const float* W2      = (const float*)d_in[4];
    const float* b2      = (const float*)d_in[5];

    float* out  = (float*)d_out;                       // logits [BT, VOCAB]
    float* loss = out + (size_t)BT * VOCAB;            // scalar loss follows logits

    // Workspace layout: W2 bf16 (25.7MB) | X bf16 (2MB) | nll (16KB)  (~28MB total)
    unsigned short* W2b = (unsigned short*)d_ws;
    unsigned short* Xb  = W2b + (size_t)VOCAB * HIDDEN;
    float*          nll = (float*)(Xb + (size_t)BT * HIDDEN);

    cvt_w2_kernel<<<VOCAB, 256, 0, stream>>>(W2, W2b);
    embed_kernel<<<BT, 256, 0, stream>>>(idx, W1, b1, Xb);

    dim3 g((VOCAB + 255) / 256, BT / 32);              // 197 x 128 workgroups
    gemm_logits_kernel<<<g, 256, 0, stream>>>(Xb, W2b, b2, out);

    lse_kernel<<<BT, 256, 0, stream>>>(out, targets, nll);
    loss_kernel<<<1, 256, 0, stream>>>(nll, loss);
}
